// S4Layer_2654289789648
// MI455X (gfx1250) — matrix operand order repaired and checked by reading, not compiled, not run on hardware
//
#include <hip/hip_runtime.h>
#include <hip/hip_bf16.h>

#define BATCH 16
#define SEQ   4096
#define DM    512
#define SD    512

typedef __attribute__((ext_vector_type(16))) __bf16 v16bf;
typedef __attribute__((ext_vector_type(8)))  __bf16 v8bf;
typedef __attribute__((ext_vector_type(8)))  float  v8f;
typedef __attribute__((ext_vector_type(4)))  float  v4f;

// ---------------------------------------------------------------------------
// Kernel 0: convert A, B, C weight matrices (each 512x512 f32) to bf16.
// ---------------------------------------------------------------------------
__global__ void s4_cvt_weights(const float* __restrict__ A,
                               const float* __restrict__ B,
                               const float* __restrict__ C,
                               __bf16* __restrict__ Ab,
                               __bf16* __restrict__ Bb,
                               __bf16* __restrict__ Cb) {
    int i = blockIdx.x * blockDim.x + threadIdx.x;   // 1024 blocks * 256 = 262144
    Ab[i] = (__bf16)A[i];
    Bb[i] = (__bf16)B[i];
    Cb[i] = (__bf16)C[i];
}

// Load a 16x32 WMMA A-operand fragment from a row-major bf16 matrix.
// Lane l: row = base row + (l&15); elems j=0..7 at k0, j=8..15 at k0+8.
__device__ __forceinline__ v16bf load_wfrag(const __bf16* row, int kt, int g) {
    int k0 = kt * 32 + g * 16;
    v8bf a0 = *(const v8bf*)(row + k0);
    v8bf a1 = *(const v8bf*)(row + k0 + 8);
    v16bf a;
#pragma unroll
    for (int j = 0; j < 8; ++j) { a[j] = a0[j]; a[j + 8] = a1[j]; }
    return a;
}

// ---------------------------------------------------------------------------
// Kernel 1: bx[t][b][n] = sum_d B[n][d] * x[b][t][d]   (bx stored bf16)
// WMMA roles: M = 16 rows of B (ntile), N = batch(16), K = d.
// One wave per (t, ntile).
// ---------------------------------------------------------------------------
__global__ void __launch_bounds__(256)
s4_input_proj(const float* __restrict__ x,
              const __bf16* __restrict__ Bb,
              __bf16* __restrict__ bx) {
    int wave  = (blockIdx.x * blockDim.x + threadIdx.x) >> 5;
    int lane  = threadIdx.x & 31;
    int t     = wave >> 5;      // [0, 4096)
    int ntile = wave & 31;      // [0, 32)
    int g = lane >> 4;          // half-wave group
    int r = lane & 15;

    const __bf16* brow = Bb + (size_t)(ntile * 16 + r) * DM;       // B[n][:]  (A-op)
    const float*  xrow = x  + ((size_t)r * SEQ + (size_t)t) * DM;  // x[b=r][t][:] (B-op)

    v8f acc = {};
    for (int kt = 0; kt < DM / 32; ++kt) {
        v16bf a = load_wfrag(brow, kt, g);
        // B-fragment: N = b = r, K = kt*32 + g*16 + j (16 contiguous f32 -> bf16)
        int kb = kt * 32 + g * 16;
        v4f x0 = *(const v4f*)(xrow + kb);
        v4f x1 = *(const v4f*)(xrow + kb + 4);
        v4f x2 = *(const v4f*)(xrow + kb + 8);
        v4f x3 = *(const v4f*)(xrow + kb + 12);
        v16bf b;
#pragma unroll
        for (int j = 0; j < 4; ++j) {
            b[j]      = (__bf16)x0[j];
            b[j + 4]  = (__bf16)x1[j];
            b[j + 8]  = (__bf16)x2[j];
            b[j + 12] = (__bf16)x3[j];
        }
        acc = __builtin_amdgcn_wmma_f32_16x16x32_bf16(false, a, false, b,
                                                      (short)0, acc, false, false);
    }
    // D: elem j -> (n = ntile*16 + g*8 + j, b = r): 8 consecutive n -> one 16B store
    __bf16* o = bx + ((size_t)t * BATCH + (size_t)r) * SD + ntile * 16 + g * 8;
    v8bf hv;
#pragma unroll
    for (int j = 0; j < 8; ++j) hv[j] = (__bf16)acc[j];
    *(v8bf*)o = hv;
}

// ---------------------------------------------------------------------------
// Kernel 2: sequential scan. ONE workgroup, 32 waves; wave w owns N-tile w
// (16 rows of A / 16 state channels). WMMA: M = A-rows, N = batch, K = state.
// State double-buffered in LDS as bf16 [b][k]; bx_t software-pipelined into
// the accumulator; dual WMMA chains halve the serial dependency depth;
// states streamed to global (bf16).
// ---------------------------------------------------------------------------
__global__ void __launch_bounds__(1024, 1)
s4_scan(const __bf16* __restrict__ bx,
        const __bf16* __restrict__ Ab,
        __bf16* __restrict__ states) {
    __shared__ __bf16 sb[2][BATCH][SD];   // 32 KB

    int tid  = threadIdx.x;
    int w    = tid >> 5;                  // ntile 0..31
    int lane = tid & 31;
    int g = lane >> 4;
    int r = lane & 15;

    // zero-init state buffer 0
    for (int i = tid; i < BATCH * SD; i += 1024) ((__bf16*)sb)[i] = (__bf16)0.0f;
    __syncthreads();

    const __bf16* arow = Ab + (size_t)(w * 16 + r) * SD;       // A[n][:] (A-operand)
    // per-lane addresses for this wave's D slice: (b = r, n = w*16 + g*8 + j)
    const size_t dcol = (size_t)r * SD + (size_t)(w * 16 + g * 8);
    const __bf16* bx_l = bx     + dcol;   // + t*BATCH*SD
    __bf16*       st_l = states + dcol;   // + t*BATCH*SD

    // prefetch bx for t = 0
    v8bf pv = *(const v8bf*)(bx_l);

#pragma unroll 2
    for (int t = 0; t < SEQ; ++t) {
        int cur = t & 1, nxt = cur ^ 1;
        // chain 0 initialized with bx_t (prefetched during previous step)
        v8f acc0, acc1 = {};
#pragma unroll
        for (int j = 0; j < 8; ++j) acc0[j] = (float)pv[j];

#pragma unroll 2
        for (int kt = 0; kt < SD / 32; kt += 2) {
            // ---- even k-tile -> chain 0
            v16bf a0 = load_wfrag(arow, kt, g);                // A slice from L2
            int kb0 = kt * 32 + g * 16;
            v8bf s0a = *(const v8bf*)&sb[cur][r][kb0];
            v8bf s0b = *(const v8bf*)&sb[cur][r][kb0 + 8];
            v16bf b0;
#pragma unroll
            for (int j = 0; j < 8; ++j) { b0[j] = s0a[j]; b0[j + 8] = s0b[j]; }
            acc0 = __builtin_amdgcn_wmma_f32_16x16x32_bf16(false, a0, false, b0,
                                                           (short)0, acc0, false, false);
            // ---- odd k-tile -> chain 1
            v16bf a1 = load_wfrag(arow, kt + 1, g);
            int kb1 = kb0 + 32;
            v8bf s1a = *(const v8bf*)&sb[cur][r][kb1];
            v8bf s1b = *(const v8bf*)&sb[cur][r][kb1 + 8];
            v16bf b1;
#pragma unroll
            for (int j = 0; j < 8; ++j) { b1[j] = s1a[j]; b1[j + 8] = s1b[j]; }
            acc1 = __builtin_amdgcn_wmma_f32_16x16x32_bf16(false, a1, false, b1,
                                                           (short)0, acc1, false, false);
        }

        // prefetch bx for next step (kept in-bounds at t = SEQ-1)
        pv = *(const v8bf*)(bx_l + (size_t)(t + 1 < SEQ ? t + 1 : t) * BATCH * SD);

        // combine chains, relu, pack to bf16: one 16B LDS store + 16B global
        v8bf hv;
#pragma unroll
        for (int j = 0; j < 8; ++j) {
            float v = acc0[j] + acc1[j];
            v = v > 0.0f ? v : 0.0f;
            hv[j] = (__bf16)v;
        }
        *(v8bf*)&sb[nxt][r][w * 16 + g * 8] = hv;
        *(v8bf*)(st_l + (size_t)t * BATCH * SD) = hv;

        __syncthreads();
    }
}

// ---------------------------------------------------------------------------
// Kernel 3: y[b][t][d] = sum_n C[d][n] * states[t][b][n]
// WMMA roles: M = 16 rows of C (dtile), N = batch(16), K = n.
// One wave per (t, dtile).
// ---------------------------------------------------------------------------
__global__ void __launch_bounds__(256)
s4_output_proj(const __bf16* __restrict__ states,
               const __bf16* __restrict__ Cb,
               float* __restrict__ out) {
    int wave  = (blockIdx.x * blockDim.x + threadIdx.x) >> 5;
    int lane  = threadIdx.x & 31;
    int t     = wave >> 5;
    int dtile = wave & 31;
    int g = lane >> 4;
    int r = lane & 15;

    const __bf16* crow = Cb + (size_t)(dtile * 16 + r) * SD;                 // C[d][:]
    const __bf16* srow = states + ((size_t)t * BATCH + (size_t)r) * SD;      // state[t][b=r][:]

    v8f acc = {};
    for (int kt = 0; kt < SD / 32; ++kt) {
        v16bf a = load_wfrag(crow, kt, g);
        int kb = kt * 32 + g * 16;
        v8bf s0 = *(const v8bf*)(srow + kb);
        v8bf s1 = *(const v8bf*)(srow + kb + 8);
        v16bf b;
#pragma unroll
        for (int j = 0; j < 8; ++j) { b[j] = s0[j]; b[j + 8] = s1[j]; }

        acc = __builtin_amdgcn_wmma_f32_16x16x32_bf16(false, a, false, b,
                                                      (short)0, acc, false, false);
    }
    // D: elem j -> (d = dtile*16 + g*8 + j, b = r): contiguous d -> 2x v4f store
    float* o = out + ((size_t)r * SEQ + (size_t)t) * DM + dtile * 16 + g * 8;
    v4f lo, hi;
#pragma unroll
    for (int j = 0; j < 4; ++j) { lo[j] = acc[j]; hi[j] = acc[j + 4]; }
    *(v4f*)o = lo;
    *(v4f*)(o + 4) = hi;
}

// ---------------------------------------------------------------------------
extern "C" void kernel_launch(void* const* d_in, const int* in_sizes, int n_in,
                              void* d_out, int out_size, void* d_ws, size_t ws_size,
                              hipStream_t stream) {
    const float* x = (const float*)d_in[0];   // [16, 4096, 512]
    const float* A = (const float*)d_in[1];   // [512, 512]
    const float* B = (const float*)d_in[2];   // [512, 512]
    const float* C = (const float*)d_in[3];   // [512, 512]
    float* out = (float*)d_out;               // [16, 4096, 512]

    char* ws = (char*)d_ws;
    const size_t seq_elems = (size_t)SEQ * BATCH * SD;        // 33554432
    __bf16* bx     = (__bf16*)ws;                             // 67 MB (bf16)
    __bf16* states = bx + seq_elems;                          // 67 MB (bf16)
    __bf16* Ab     = states + seq_elems;
    __bf16* Bb     = Ab + (size_t)SD * SD;
    __bf16* Cb     = Bb + (size_t)SD * DM;

    // Phase 0: weight conversion (262144 elems each)
    s4_cvt_weights<<<1024, 256, 0, stream>>>(A, B, C, Ab, Bb, Cb);

    // Phase 1: bx = B @ x_t  (4096*32 wave-tiles, 8 waves/block)
    s4_input_proj<<<(SEQ * 32) / 8, 256, 0, stream>>>(x, Bb, bx);

    // Phase 2: sequential recurrence on a single WGP
    s4_scan<<<1, 1024, 0, stream>>>(bx, Ab, states);

    // Phase 3: y = C @ state_t
    s4_output_proj<<<(SEQ * 32) / 8, 256, 0, stream>>>(states, Cb, out);
}
